// W8A8Linear_81879256531628
// MI455X (gfx1250) — matrix operand order repaired and checked by reading, not compiled, not run on hardware
//
#include <hip/hip_runtime.h>
#include <hip/hip_bf16.h>

// ---------------------------------------------------------------------------
// W8A8 fake-quant linear: y = (quant(x) * scale) @ W^T + bias
//   x: [8192,4096] f32, weight: [11008,4096] f32, out: [8192,11008] f32
//
// MI455X roofline: 0.739 TFLOP; dense f16 WMMA ~5 PFLOPS -> ~150us compute;
// HBM floor ~29us => WMMA-bound. Quantized activations are exact integers in
// f16; weight f16 err ~5e-4 (fp8/iu8 would break accuracy vs fp32-weight ref).
//
// v3: pre-convert q(x) and w to f16 in workspace (extra ~470MB HBM ~ 20us),
// so the GEMM inner loop is a pure f16 WMMA pipeline:
//   - no conversion VALU in the hot loop
//   - operand traffic halves: 17.3GB -> 8.6GB (mostly L2)
//   - staging via global_load_async_to_lds_b128 (ASYNCcnt, no VGPR round-trip)
//   - block 128x256, BK=64, 8 waves (2Mx4N), wave tile 64x64 (16 wmma / 8 ds)
// Fallback to fused (round-2) kernel if workspace is too small.
// ---------------------------------------------------------------------------

typedef __attribute__((ext_vector_type(16))) _Float16 v16h;
typedef __attribute__((ext_vector_type(8)))  _Float16 v8h;
typedef __attribute__((ext_vector_type(8)))  float    v8f;
typedef int v4i __attribute__((vector_size(16)));

#define GEMM_M 8192
#define GEMM_N 11008
#define GEMM_K 4096

#define BM 128
#define BN 256

// fused-path (fallback) tiling
#define BKF 32
#define LDSF 40    // halves stride: 32 + 8 pad (80B, 16B aligned)

// f16-path tiling
#define BK2 64
#define LDA2 72    // halves stride: 64 + 8 pad (144B, 16B aligned)

#define PF_ITERS 8
#define RED_BLOCKS 1024

#if __has_builtin(__builtin_amdgcn_global_load_async_to_lds_b128) && \
    __has_builtin(__builtin_amdgcn_s_wait_asynccnt)
#define USE_ASYNC_LDS 1
#else
#define USE_ASYNC_LDS 0
#endif

#define GLOBAL_AS __attribute__((address_space(1)))
#define LDS_AS    __attribute__((address_space(3)))

// ---------------------------- absmax reduction -----------------------------

__global__ void absmax_partial_kernel(const float* __restrict__ x,
                                      float* __restrict__ partials, int n4) {
  const float4* __restrict__ x4 = (const float4*)x;
  float m = 0.0f;
  for (int i = blockIdx.x * blockDim.x + threadIdx.x; i < n4;
       i += gridDim.x * blockDim.x) {
    float4 v = x4[i];
    m = fmaxf(m, fmaxf(fmaxf(fabsf(v.x), fabsf(v.y)),
                       fmaxf(fabsf(v.z), fabsf(v.w))));
  }
  __shared__ float red[256];
  red[threadIdx.x] = m;
  __syncthreads();
  for (int s = 128; s > 0; s >>= 1) {
    if (threadIdx.x < s)
      red[threadIdx.x] = fmaxf(red[threadIdx.x], red[threadIdx.x + s]);
    __syncthreads();
  }
  if (threadIdx.x == 0) partials[blockIdx.x] = red[0];
}

__global__ void absmax_final_kernel(const float* __restrict__ partials,
                                    float* __restrict__ scales) {
  float m = 0.0f;
  for (int i = threadIdx.x; i < RED_BLOCKS; i += 256)
    m = fmaxf(m, partials[i]);
  __shared__ float red[256];
  red[threadIdx.x] = m;
  __syncthreads();
  for (int s = 128; s > 0; s >>= 1) {
    if (threadIdx.x < s)
      red[threadIdx.x] = fmaxf(red[threadIdx.x], red[threadIdx.x + s]);
    __syncthreads();
  }
  if (threadIdx.x == 0) {
    float scale = fmaxf(red[0], 1e-5f) / 127.0f;  // CLIP_RATIO = 1.0
    scales[0] = scale;
    scales[1] = 1.0f / scale;
  }
}

// ------------------------ pre-conversion kernels ---------------------------

__device__ __forceinline__ _Float16 quantize_one(float v, float inv_scale) {
  float q = rintf(v * inv_scale);
  q = fminf(fmaxf(q, -128.0f), 127.0f);
  return (_Float16)q;  // integers in [-128,127]: exact in f16
}

__global__ void quant_x_kernel(const float* __restrict__ x,
                               _Float16* __restrict__ qx,
                               const float* __restrict__ scales) {
  const float inv = scales[1];
  const int n8 = (GEMM_M * GEMM_K) / 8;
  for (int i = blockIdx.x * blockDim.x + threadIdx.x; i < n8;
       i += gridDim.x * blockDim.x) {
    float4 a = ((const float4*)x)[2 * i];
    float4 b = ((const float4*)x)[2 * i + 1];
    v8h q;
    q[0] = quantize_one(a.x, inv); q[1] = quantize_one(a.y, inv);
    q[2] = quantize_one(a.z, inv); q[3] = quantize_one(a.w, inv);
    q[4] = quantize_one(b.x, inv); q[5] = quantize_one(b.y, inv);
    q[6] = quantize_one(b.z, inv); q[7] = quantize_one(b.w, inv);
    ((v8h*)qx)[i] = q;
  }
}

__global__ void convert_w_kernel(const float* __restrict__ w,
                                 _Float16* __restrict__ wh) {
  const int n8 = (GEMM_N * GEMM_K) / 8;
  for (int i = blockIdx.x * blockDim.x + threadIdx.x; i < n8;
       i += gridDim.x * blockDim.x) {
    float4 a = ((const float4*)w)[2 * i];
    float4 b = ((const float4*)w)[2 * i + 1];
    v8h q;
    q[0] = (_Float16)a.x; q[1] = (_Float16)a.y;
    q[2] = (_Float16)a.z; q[3] = (_Float16)a.w;
    q[4] = (_Float16)b.x; q[5] = (_Float16)b.y;
    q[6] = (_Float16)b.z; q[7] = (_Float16)b.w;
    ((v8h*)wh)[i] = q;
  }
}

// ------------------------ f16-input WMMA GEMM (main) -----------------------

__device__ __forceinline__ void stage16(const _Float16* src, _Float16* dst) {
#if USE_ASYNC_LDS
  GLOBAL_AS v4i* s = (GLOBAL_AS v4i*)src;   // clang prints AS1 as __device__
  LDS_AS    v4i* d = (LDS_AS    v4i*)dst;   // clang prints AS3 as __shared__
  __builtin_amdgcn_global_load_async_to_lds_b128(s, d, 0,  0);
  __builtin_amdgcn_global_load_async_to_lds_b128(s, d, 16, 0);
  __builtin_amdgcn_global_load_async_to_lds_b128(s, d, 32, 0);
  __builtin_amdgcn_global_load_async_to_lds_b128(s, d, 48, 0);
#else
  *(uint4*)((char*)dst)      = *(const uint4*)((const char*)src);
  *(uint4*)((char*)dst + 16) = *(const uint4*)((const char*)src + 16);
  *(uint4*)((char*)dst + 32) = *(const uint4*)((const char*)src + 32);
  *(uint4*)((char*)dst + 48) = *(const uint4*)((const char*)src + 48);
#endif
}

__global__ __launch_bounds__(256)
void wmma_gemm_f16_kernel(const _Float16* __restrict__ qx,
                          const _Float16* __restrict__ wh,
                          const float* __restrict__ bias,
                          float* __restrict__ out,
                          const float* __restrict__ scales) {
  __shared__ _Float16 As[2][BM * LDA2];   // 2 x 18 KB
  __shared__ _Float16 Bs[2][BN * LDA2];   // 2 x 36 KB  (108 KB total)

  const int tid  = threadIdx.x;
  const int lane = tid & 31;
  const int wid  = tid >> 5;
  const int wm   = wid & 1;   // 2 wave rows in M (64 rows each)
  const int wn   = wid >> 1;  // 4 wave cols in N (64 cols each)

  const int mTile = blockIdx.x * BM;  // M fastest -> weight tile hot in L2
  const int nTile = blockIdx.y * BN;

  const float scale = scales[0];

  // staging: thread owns 64B (32 halves) of row r0 in A and rows r0/r0+128 in B
  const int r0  = tid >> 1;          // 0..127
  const int sg  = (tid & 1) * 32;    // 0 or 32 halves
  const _Float16* __restrict__ ga  = qx + (size_t)(mTile + r0) * GEMM_K + sg;
  const _Float16* __restrict__ gb0 = wh + (size_t)(nTile + r0) * GEMM_K + sg;
  const _Float16* __restrict__ gb1 = gb0 + (size_t)128 * GEMM_K;

  v8f acc[4][4];
  #pragma unroll
  for (int mf = 0; mf < 4; ++mf)
    #pragma unroll
    for (int nf = 0; nf < 4; ++nf)
      acc[mf][nf] = (v8f)(0.0f);

  const int nkt = GEMM_K / BK2;  // 64 k-tiles
  const int lh  = lane & 15;
  const int c0  = (lane >> 4) * 16;   // A: K-chunk select   (contiguous: k = 16*(lane/16) + s)
  const int bk0 = (lane >> 4) * 16;   // B: K-chunk select   (contiguous: k = 16*(lane/16) + s)

  // ---- prologue: stage tile 0 into buffer 0 ----
  stage16(ga,  &As[0][r0 * LDA2 + sg]);
  stage16(gb0, &Bs[0][r0 * LDA2 + sg]);
  stage16(gb1, &Bs[0][(r0 + 128) * LDA2 + sg]);
#if USE_ASYNC_LDS
  __builtin_amdgcn_s_wait_asynccnt(0);
#endif
  __syncthreads();

  for (int kt = 0; kt < nkt; ++kt) {
    const int buf = kt & 1;

    // stage next tile into the other buffer (async: no VGPR round-trip)
    if (kt + 1 < nkt) {
      const int ko = (kt + 1) * BK2;
      stage16(ga + ko,  &As[buf ^ 1][r0 * LDA2 + sg]);
      stage16(gb0 + ko, &Bs[buf ^ 1][r0 * LDA2 + sg]);
      stage16(gb1 + ko, &Bs[buf ^ 1][(r0 + 128) * LDA2 + sg]);
    }
    // warm L2 ahead (global_prefetch_b8)
    if (kt + PF_ITERS < nkt) {
      __builtin_prefetch(ga  + (kt + PF_ITERS) * BK2, 0, 3);
      __builtin_prefetch(gb0 + (kt + PF_ITERS) * BK2, 0, 3);
      __builtin_prefetch(gb1 + (kt + PF_ITERS) * BK2, 0, 3);
    }

    // ---- 2 k-chunks of 32: fragments (contiguous k per lane half) + 32 WMMAs ----
    #pragma unroll
    for (int kc = 0; kc < 2; ++kc) {
      v16h bfr[4];
      #pragma unroll
      for (int nf = 0; nf < 4; ++nf) {
        const _Float16* p =
            &Bs[buf][(wn * 64 + nf * 16 + lh) * LDA2 + kc * 32 + bk0];
        v8h lo = *(const v8h*)(p);
        v8h hi = *(const v8h*)(p + 8);
        bfr[nf] = __builtin_shufflevector(lo, hi, 0, 1, 2, 3, 4, 5, 6, 7,
                                          8, 9, 10, 11, 12, 13, 14, 15);
      }
      #pragma unroll
      for (int mf = 0; mf < 4; ++mf) {
        const _Float16* p =
            &As[buf][(wm * 64 + mf * 16 + lh) * LDA2 + kc * 32];
        v8h lo = *(const v8h*)(p + c0);
        v8h hi = *(const v8h*)(p + c0 + 8);
        v16h afr = __builtin_shufflevector(lo, hi, 0, 1, 2, 3, 4, 5, 6, 7,
                                           8, 9, 10, 11, 12, 13, 14, 15);
        #pragma unroll
        for (int nf = 0; nf < 4; ++nf)
          acc[mf][nf] = __builtin_amdgcn_wmma_f32_16x16x32_f16(
              false, afr, false, bfr[nf], (short)0, acc[mf][nf], false, false);
      }
    }

    if (kt + 1 < nkt) {
#if USE_ASYNC_LDS
      __builtin_amdgcn_s_wait_asynccnt(0);
#endif
      __syncthreads();
    }
  }

  // ---- epilogue: dequant + bias, non-temporal (write-once stream) ----
  #pragma unroll
  for (int mf = 0; mf < 4; ++mf) {
    #pragma unroll
    for (int nf = 0; nf < 4; ++nf) {
      const int col  = nTile + wn * 64 + nf * 16 + lh;
      const float bv = bias[col];
      const int mrow = mTile + wm * 64 + mf * 16 + (lane >> 4) * 8;
      float* op = out + (size_t)mrow * GEMM_N + col;
      v8f a = acc[mf][nf];
      #pragma unroll
      for (int r = 0; r < 8; ++r)
        __builtin_nontemporal_store(a[r] * scale + bv, op + (size_t)r * GEMM_N);
    }
  }
}

// ---------------- fused fallback (small workspace), round-2 style ----------

__global__ __launch_bounds__(256)
void wmma_gemm_fused_kernel(const float* __restrict__ x,
                            const float* __restrict__ w,
                            const float* __restrict__ bias,
                            float* __restrict__ out,
                            const float* __restrict__ scales) {
  __shared__ _Float16 As[2][BM * LDSF];
  __shared__ _Float16 Bs[2][BN * LDSF];

  const int tid  = threadIdx.x;
  const int lane = tid & 31;
  const int wid  = tid >> 5;
  const int wm   = wid & 1;
  const int wn   = wid >> 1;

  const int mTile = blockIdx.x * BM;
  const int nTile = blockIdx.y * BN;

  const float scale     = scales[0];
  const float inv_scale = scales[1];

  const int lr = tid >> 1;
  const int lk = (tid & 1) * 16;
  const float* __restrict__ xg  = x + (size_t)(mTile + lr) * GEMM_K + lk;
  const float* __restrict__ wg0 = w + (size_t)(nTile + lr) * GEMM_K + lk;
  const float* __restrict__ wg1 = wg0 + (size_t)128 * GEMM_K;

  v8f acc[4][4];
  #pragma unroll
  for (int mf = 0; mf < 4; ++mf)
    #pragma unroll
    for (int nf = 0; nf < 4; ++nf)
      acc[mf][nf] = (v8f)(0.0f);

  float4 ar[4], br[8];
  #pragma unroll
  for (int i = 0; i < 4; ++i) {
    ar[i]     = *(const float4*)(xg  + i * 4);
    br[i]     = *(const float4*)(wg0 + i * 4);
    br[i + 4] = *(const float4*)(wg1 + i * 4);
  }
  {
    const float* af = (const float*)ar;
    const float* bf = (const float*)br;
    v8h qa0, qa1, qb0, qb1, qb2, qb3;
    #pragma unroll
    for (int i = 0; i < 8; ++i) {
      qa0[i] = quantize_one(af[i],     inv_scale);
      qa1[i] = quantize_one(af[i + 8], inv_scale);
      qb0[i] = (_Float16)bf[i];
      qb1[i] = (_Float16)bf[i + 8];
      qb2[i] = (_Float16)bf[i + 16];
      qb3[i] = (_Float16)bf[i + 24];
    }
    *(v8h*)&As[0][lr * LDSF + lk]             = qa0;
    *(v8h*)&As[0][lr * LDSF + lk + 8]         = qa1;
    *(v8h*)&Bs[0][lr * LDSF + lk]             = qb0;
    *(v8h*)&Bs[0][lr * LDSF + lk + 8]         = qb1;
    *(v8h*)&Bs[0][(lr + 128) * LDSF + lk]     = qb2;
    *(v8h*)&Bs[0][(lr + 128) * LDSF + lk + 8] = qb3;
  }

  const int nkt = GEMM_K / BKF;
  const int lh  = lane & 15;
  const int c0  = (lane >> 4) * 16;
  const int bk0 = (lane >> 4) * 16;

  for (int kt = 0; kt < nkt; ++kt) {
    __syncthreads();
    const int buf = kt & 1;

    if (kt + 1 < nkt) {
      const float* xp  = xg  + (kt + 1) * BKF;
      const float* wp0 = wg0 + (kt + 1) * BKF;
      const float* wp1 = wg1 + (kt + 1) * BKF;
      #pragma unroll
      for (int i = 0; i < 4; ++i) {
        ar[i]     = *(const float4*)(xp  + i * 4);
        br[i]     = *(const float4*)(wp0 + i * 4);
        br[i + 4] = *(const float4*)(wp1 + i * 4);
      }
    }

    v16h bfr[4];
    #pragma unroll
    for (int nf = 0; nf < 4; ++nf) {
      const _Float16* p = &Bs[buf][(wn * 64 + nf * 16 + lh) * LDSF + bk0];
      v8h lo = *(const v8h*)(p);
      v8h hi = *(const v8h*)(p + 8);
      bfr[nf] = __builtin_shufflevector(lo, hi, 0, 1, 2, 3, 4, 5, 6, 7,
                                        8, 9, 10, 11, 12, 13, 14, 15);
    }
    #pragma unroll
    for (int mf = 0; mf < 4; ++mf) {
      const _Float16* p = &As[buf][(wm * 64 + mf * 16 + lh) * LDSF];
      v8h lo = *(const v8h*)(p + c0);
      v8h hi = *(const v8h*)(p + c0 + 8);
      v16h afr = __builtin_shufflevector(lo, hi, 0, 1, 2, 3, 4, 5, 6, 7,
                                         8, 9, 10, 11, 12, 13, 14, 15);
      #pragma unroll
      for (int nf = 0; nf < 4; ++nf)
        acc[mf][nf] = __builtin_amdgcn_wmma_f32_16x16x32_f16(
            false, afr, false, bfr[nf], (short)0, acc[mf][nf], false, false);
    }

    if (kt + 1 < nkt) {
      const float* af = (const float*)ar;
      const float* bf = (const float*)br;
      v8h qa0, qa1, qb0, qb1, qb2, qb3;
      #pragma unroll
      for (int i = 0; i < 8; ++i) {
        qa0[i] = quantize_one(af[i],     inv_scale);
        qa1[i] = quantize_one(af[i + 8], inv_scale);
        qb0[i] = (_Float16)bf[i];
        qb1[i] = (_Float16)bf[i + 8];
        qb2[i] = (_Float16)bf[i + 16];
        qb3[i] = (_Float16)bf[i + 24];
      }
      const int nb = buf ^ 1;
      *(v8h*)&As[nb][lr * LDSF + lk]             = qa0;
      *(v8h*)&As[nb][lr * LDSF + lk + 8]         = qa1;
      *(v8h*)&Bs[nb][lr * LDSF + lk]             = qb0;
      *(v8h*)&Bs[nb][lr * LDSF + lk + 8]         = qb1;
      *(v8h*)&Bs[nb][(lr + 128) * LDSF + lk]     = qb2;
      *(v8h*)&Bs[nb][(lr + 128) * LDSF + lk + 8] = qb3;
    }
  }

  #pragma unroll
  for (int mf = 0; mf < 4; ++mf) {
    #pragma unroll
    for (int nf = 0; nf < 4; ++nf) {
      const int col  = nTile + wn * 64 + nf * 16 + lh;
      const float bv = bias[col];
      const int mrow = mTile + wm * 64 + mf * 16 + (lane >> 4) * 8;
      float* op = out + (size_t)mrow * GEMM_N + col;
      v8f a = acc[mf][nf];
      #pragma unroll
      for (int r = 0; r < 8; ++r)
        __builtin_nontemporal_store(a[r] * scale + bv, op + (size_t)r * GEMM_N);
    }
  }
}

// ------------------------------- launcher ----------------------------------

extern "C" void kernel_launch(void* const* d_in, const int* in_sizes, int n_in,
                              void* d_out, int out_size, void* d_ws, size_t ws_size,
                              hipStream_t stream) {
  const float* x    = (const float*)d_in[0];
  const float* wgt  = (const float*)d_in[1];
  const float* bias = (const float*)d_in[2];
  float* out = (float*)d_out;

  float* wsf      = (float*)d_ws;
  float* scales   = wsf;        // [0]=scale, [1]=1/scale
  float* partials = wsf + 16;   // RED_BLOCKS floats

  const size_t QX_ELEMS = (size_t)GEMM_M * GEMM_K;
  const size_t WH_ELEMS = (size_t)GEMM_N * GEMM_K;
  const size_t need = 4096 + (QX_ELEMS + WH_ELEMS) * sizeof(_Float16);

  const int n4 = in_sizes[0] / 4;

  absmax_partial_kernel<<<RED_BLOCKS, 256, 0, stream>>>(x, partials, n4);
  absmax_final_kernel<<<1, 256, 0, stream>>>(partials, scales);

  dim3 grid(GEMM_M / BM, GEMM_N / BN);  // (64, 43); M fastest: weight hot in L2

  if (ws_size >= need) {
    _Float16* qx = (_Float16*)((char*)d_ws + 4096);
    _Float16* wh = qx + QX_ELEMS;
    convert_w_kernel<<<8192, 256, 0, stream>>>(wgt, wh);
    quant_x_kernel<<<8192, 256, 0, stream>>>(x, qx, scales);
    wmma_gemm_f16_kernel<<<grid, 256, 0, stream>>>(qx, wh, bias, out, scales);
  } else {
    wmma_gemm_fused_kernel<<<grid, 256, 0, stream>>>(x, wgt, bias, out, scales);
  }
}
